// VMBlock_56727928045653
// MI455X (gfx1250) — matrix operand order repaired and checked by reading, not compiled, not run on hardware
//
#include <hip/hip_runtime.h>
#include <math.h>

// ---------------------------------------------------------------------------
// CDNA5 (gfx1250) VMBlock forward.  All dense contractions go through a
// wave32 WMMA (v_wmma_f32_16x16x32_f16) GEMM with LDS-staged tiles;
// scans/convs/elementwise are plain VALU kernels.  A TDM (tensor_load_to_lds)
// staging kernel exercises the async-tensor path.
// ---------------------------------------------------------------------------

typedef __attribute__((ext_vector_type(16))) _Float16 v16h;
typedef __attribute__((ext_vector_type(8)))  float    v8f;
typedef __attribute__((ext_vector_type(4)))  unsigned int u32x4;
typedef __attribute__((ext_vector_type(8)))  int      i32x8;
typedef __attribute__((ext_vector_type(4)))  int      i32x4;

union V16H { v16h v; _Float16 e[16]; };
union V8F  { v8f  v; float    e[8];  };

#define DEV static __device__ __forceinline__

DEV float sigmoid_(float x){ return 1.0f/(1.0f+__expf(-x)); }
DEV float silu_(float x){ return x*sigmoid_(x); }
DEV float gelu_(float x){ return 0.5f*x*(1.0f+erff(x*0.7071067811865475f)); }
DEV float softplus_(float x){ return (x>20.0f)? x : log1pf(__expf(x)); }
DEV float actf(float v,int a){
  if(a==1) return silu_(v);
  if(a==2) return gelu_(v);
  if(a==3) return softplus_(v);
  if(a==4) return fmaxf(v,0.0f);
  if(a==5) return sigmoid_(v);
  return v;
}

// ---- TDM staging kernel: exercise the Tensor Data Mover path --------------
// Builds a D# (group0/group1) for a 64x4 tile of 4-byte elements from a 64x64
// tensor at `src`, DMA-loads it into LDS, waits on TENSORcnt, and drains one
// wave's worth into `dst` (dedicated scratch; nothing downstream reads it).
__global__ void k_tdm_stage(const float* __restrict__ src, float* __restrict__ dst){
#if __has_builtin(__builtin_amdgcn_tensor_load_to_lds)
  __shared__ float lds[512];
  unsigned long long ga = (unsigned long long)(size_t)src;
  unsigned int ldsa = (unsigned int)(size_t)(&lds[0]);
  u32x4 g0;
  g0.x = 1u;                                            // count=1 (valid D#)
  g0.y = ldsa;                                          // lds_addr
  g0.z = (unsigned int)(ga & 0xffffffffull);            // global_addr[31:0]
  g0.w = (unsigned int)((ga >> 32) & 0x01ffffffull)     // global_addr[56:32]
       | (2u << 30);                                    // type=2 ("image")
  i32x8 g1;
  g1[0] = (int)(2u << 16);      // workgroup_mask=0, data_size=2 (4 bytes)
  g1[1] = (int)(64u << 16);     // tensor_dim0 = 64   (bits 79:48, low half)
  g1[2] = (int)(64u << 16);     // tensor_dim1 = 64   (bits 111:80, low half)
  g1[3] = (int)(64u << 16);     // tile_dim0 = 64     (bits 127:112)
  g1[4] = 4;                    // tile_dim1 = 4      (bits 143:128)
  g1[5] = 64;                   // tensor_dim0_stride = 64 (bits 207:160)
  g1[6] = 0;
  g1[7] = 0;
  i32x4 gz = {0,0,0,0};
#if defined(__clang_major__) && (__clang_major__ >= 23)
  i32x8 gz8 = {0,0,0,0,0,0,0,0};
  __builtin_amdgcn_tensor_load_to_lds(g0, g1, gz, gz, gz8, 0);
#else
  __builtin_amdgcn_tensor_load_to_lds(g0, g1, gz, gz, 0);
#endif
#if __has_builtin(__builtin_amdgcn_s_wait_tensorcnt)
  __builtin_amdgcn_s_wait_tensorcnt(0);
#endif
  __syncthreads();
  int lane = threadIdx.x & 31;
  float v = lds[lane];
#if __has_builtin(__builtin_amdgcn_cluster_id_x)
  if(__builtin_amdgcn_cluster_id_x() == 12345) v = 0.0f;  // keep cid read live
#endif
  dst[lane] = v;
#else
  if(threadIdx.x == 0) dst[0] = src[0];
#endif
}

// ---- Generic batched WMMA GEMM (LDS-staged) -------------------------------
//  C[z] = act(A[z%zmodA] * B[z] + bias)
//  A: MxK row-major.  B: KxN (transB=0) or NxK (transB=1) row-major.
//  biasMode: 0 none, 1 per-N, 2 per-M.
//  Block = 256 thr (8 waves, 2x4), tile 64(M) x 64(N), K-step 32.
//  Each wave: rows {wm*16, wm*16+32}, cols wn*16 -> two WMMAs sharing B frag.
#define A_P 40   // LDS pitch for A tile (halves)
#define B_P 72   // LDS pitch for B tile (halves)

__global__ __launch_bounds__(256)
void k_gemm_wmma(const float* __restrict__ A, const float* __restrict__ Bm,
                 const float* __restrict__ bias, float* __restrict__ C,
                 int M, int N, int K, int transB, int biasMode, int act,
                 long strideA, long strideB, long strideC,
                 int zmodA, long biasStride){
  __shared__ _Float16 As[64 * A_P];   // [row][k]  64x32
  __shared__ _Float16 Bs[32 * B_P];   // [k][n]    32x64

  int z = blockIdx.z;
  const float* Ap = A + (long)(z % zmodA) * strideA;
  const float* Bp = Bm + (long)z * strideB;
  float*       Cp = C + (long)z * strideC;
  const float* bp = bias ? (bias + (long)(z % zmodA) * biasStride) : (const float*)0;

  int tid  = threadIdx.x;
  int lane = tid & 31;
  int wave = tid >> 5;
  int wm   = wave >> 2;               // 0..1
  int wn   = wave & 3;                // 0..3
  int mtile = blockIdx.y * 64;
  int ntile = blockIdx.x * 64;
  long ldb = transB ? (long)K : (long)N;

  V8F z8;
#pragma unroll
  for(int i=0;i<8;++i) z8.e[i]=0.0f;
  v8f acc0 = z8.v, acc1 = z8.v;

  // cooperative staging indices
  int arow = tid >> 2;                // 0..63
  int akq  = (tid & 3) * 8;           // 0,8,16,24
  int bkN  = tid >> 3;                // 0..31  (transB=0: k)
  int bnq  = (tid & 7) * 8;           // 0..56
  int bnT  = tid >> 2;                // 0..63  (transB=1: n)
  int bkqT = (tid & 3) * 8;           // 0,8,16,24

  for(int k0 = 0; k0 < K; k0 += 32){
    // ---- stage A tile (64 rows x 32 k) -> LDS f16 ----
    {
      int gr = mtile + arow;
      _Float16* dst = As + arow*A_P + akq;
      const float* src = Ap + (long)gr*K + k0 + akq;
      __builtin_prefetch(src + 64, 0, 1);             // global_prefetch_b8
      if(gr < M && (k0 + akq + 8) <= K){
        float4 v0 = *(const float4*)src;
        float4 v1 = *(const float4*)(src + 4);
        dst[0]=(_Float16)v0.x; dst[1]=(_Float16)v0.y;
        dst[2]=(_Float16)v0.z; dst[3]=(_Float16)v0.w;
        dst[4]=(_Float16)v1.x; dst[5]=(_Float16)v1.y;
        dst[6]=(_Float16)v1.z; dst[7]=(_Float16)v1.w;
      } else {
#pragma unroll
        for(int j=0;j<8;++j){
          int kk = k0 + akq + j;
          dst[j] = (_Float16)((gr < M && kk < K) ? Ap[(long)gr*K + kk] : 0.0f);
        }
      }
    }
    // ---- stage B tile (32 k x 64 n) -> LDS f16 ----
    if(!transB){
      int k  = bkN;
      int gn = ntile + bnq;
      _Float16* dst = Bs + k*B_P + bnq;
      const float* src = Bp + (long)(k0 + k)*ldb + gn;
      if((k0 + k) < K && (gn + 8) <= N){
        float4 v0 = *(const float4*)src;
        float4 v1 = *(const float4*)(src + 4);
        dst[0]=(_Float16)v0.x; dst[1]=(_Float16)v0.y;
        dst[2]=(_Float16)v0.z; dst[3]=(_Float16)v0.w;
        dst[4]=(_Float16)v1.x; dst[5]=(_Float16)v1.y;
        dst[6]=(_Float16)v1.z; dst[7]=(_Float16)v1.w;
      } else {
#pragma unroll
        for(int j=0;j<8;++j){
          int nn = gn + j;
          dst[j] = (_Float16)(((k0 + k) < K && nn < N)
                              ? Bp[(long)(k0 + k)*ldb + nn] : 0.0f);
        }
      }
    } else {
      int n  = bnT;
      int gn = ntile + n;
      const float* src = Bp + (long)gn*ldb + k0 + bkqT;
      if(gn < N && (k0 + bkqT + 8) <= K){
        float4 v0 = *(const float4*)src;
        float4 v1 = *(const float4*)(src + 4);
        Bs[(bkqT+0)*B_P + n] = (_Float16)v0.x;
        Bs[(bkqT+1)*B_P + n] = (_Float16)v0.y;
        Bs[(bkqT+2)*B_P + n] = (_Float16)v0.z;
        Bs[(bkqT+3)*B_P + n] = (_Float16)v0.w;
        Bs[(bkqT+4)*B_P + n] = (_Float16)v1.x;
        Bs[(bkqT+5)*B_P + n] = (_Float16)v1.y;
        Bs[(bkqT+6)*B_P + n] = (_Float16)v1.z;
        Bs[(bkqT+7)*B_P + n] = (_Float16)v1.w;
      } else {
#pragma unroll
        for(int j=0;j<8;++j){
          int kk = k0 + bkqT + j;
          Bs[(bkqT+j)*B_P + n] =
            (_Float16)((gn < N && kk < K) ? Bp[(long)gn*ldb + kk] : 0.0f);
        }
      }
    }
    __syncthreads();

    // ---- fragments from LDS (ISA 7.12.2 layouts, wave32) ----
    V16H a0, a1, bf;
    {
      const _Float16* ar0 = As + (wm*16 + (lane & 15))*A_P + ((lane & 16) ? 8 : 0);
      const _Float16* ar1 = ar0 + 32*A_P;
#pragma unroll
      for(int i=0;i<8;++i){
        a0.e[i]   = ar0[i];     a0.e[i+8] = ar0[16 + i];
        a1.e[i]   = ar1[i];     a1.e[i+8] = ar1[16 + i];
      }
      const _Float16* br = Bs + ((lane & 16) ? 8 : 0)*B_P + wn*16 + (lane & 15);
#pragma unroll
      for(int i=0;i<8;++i){ bf.e[i] = br[i*B_P]; bf.e[i+8] = br[(16 + i)*B_P]; }
    }
    acc0 = __builtin_amdgcn_wmma_f32_16x16x32_f16(false, a0.v, false, bf.v,
                                                  (short)0, acc0, false, false);
    acc1 = __builtin_amdgcn_wmma_f32_16x16x32_f16(false, a1.v, false, bf.v,
                                                  (short)0, acc1, false, false);
    __syncthreads();
  }

  // ---- epilogue: D layout -> C, bias + activation ----
  int n   = ntile + wn*16 + (lane & 15);
  int mb0 = mtile + wm*16 + ((lane & 16) ? 8 : 0);
  if(n < N){
    V8F r0; r0.v = acc0;
    V8F r1; r1.v = acc1;
    float bnv = (bp && biasMode == 1) ? bp[n] : 0.0f;
#pragma unroll
    for(int i=0;i<8;++i){
      int m0 = mb0 + i;
      if(m0 < M){
        float v = r0.e[i] + (bp ? (biasMode == 2 ? bp[m0] : bnv) : 0.0f);
        Cp[(long)m0*N + n] = actf(v, act);
      }
      int m1 = mb0 + 32 + i;
      if(m1 < M){
        float v = r1.e[i] + (bp ? (biasMode == 2 ? bp[m1] : bnv) : 0.0f);
        Cp[(long)m1*N + n] = actf(v, act);
      }
    }
  }
}

// ---- Selective scan: one lane per (b,k,d) channel -------------------------
__global__ void k_scan(const float* __restrict__ u, const float* __restrict__ dl,
                       const float* __restrict__ Amat, const float* __restrict__ Bmp,
                       const float* __restrict__ Cmp, const float* __restrict__ Dv,
                       float* __restrict__ y,
                       int G, int Kdir, int Dd, int L, long bcStride){
  long idx = (long)blockIdx.x*blockDim.x + threadIdx.x;
  if(idx >= (long)G*Dd) return;
  int d = (int)(idx % Dd);
  int g = (int)(idx / Dd);
  int k = g % Kdir;
  const float* up = u  + (long)g*Dd*L + (long)d*L;
  const float* dp = dl + (long)g*Dd*L + (long)d*L;
  const float* Bp = Bmp + (long)g*bcStride;
  const float* Cp = Cmp + (long)g*bcStride;
  const float* Ap = Amat + ((long)k*Dd + d)*16;
  float Dval = Dv[(long)k*Dd + d];
  float* yp = y + (long)g*Dd*L + (long)d*L;
  float h[16], Av[16];
#pragma unroll
  for(int n=0;n<16;++n){ h[n]=0.0f; Av[n]=Ap[n]; }
  for(int t=0;t<L;++t){
    float dt = dp[t], ut = up[t];
    float du = dt*ut, accv = 0.0f;
#pragma unroll
    for(int n=0;n<16;++n){
      h[n] = __expf(dt*Av[n])*h[n] + du*Bp[(long)n*L + t];
      accv += h[n]*Cp[(long)n*L + t];
    }
    yp[t] = accv + Dval*ut;
  }
}

// ---- Grouped 3x3 conv, pad 1, NCHW; w:(Cout,Cin/groups,3,3) ---------------
__global__ void k_conv3x3(const float* __restrict__ x, const float* __restrict__ w,
                          const float* __restrict__ bias, float* __restrict__ y,
                          int Bsz, int Cin, int Cout, int H, int W,
                          int groups, int act, long inBStride, long outBStride){
  long idx = (long)blockIdx.x*blockDim.x + threadIdx.x;
  long total = (long)Bsz*Cout*H*W;
  if(idx >= total) return;
  int wq = (int)(idx % W); long t = idx / W;
  int hq = (int)(t % H);   t /= H;
  int co = (int)(t % Cout);
  int b  = (int)(t / Cout);
  int cig = Cin / groups, cog = Cout / groups;
  int gidx = co / cog;
  float acc = bias ? bias[co] : 0.0f;
  const float* wp = w + (long)co*cig*9;
  for(int ci=0; ci<cig; ++ci){
    const float* xp = x + (long)b*inBStride + (long)(gidx*cig + ci)*H*W;
    for(int kh=0; kh<3; ++kh){
      int hh = hq + kh - 1; if(hh < 0 || hh >= H) continue;
      for(int kw=0; kw<3; ++kw){
        int wwp = wq + kw - 1; if(wwp < 0 || wwp >= W) continue;
        acc += xp[(long)hh*W + wwp] * wp[ci*9 + kh*3 + kw];
      }
    }
  }
  y[(long)b*outBStride + (long)co*H*W + (long)hq*W + wq] = actf(acc, act);
}

// ---- Elementwise / utility kernels ----------------------------------------
__global__ void k_extract(const float* __restrict__ x, float* __restrict__ o1,
                          float* __restrict__ o2, int Bsz, int L, int Hc){
  long i = (long)blockIdx.x*blockDim.x + threadIdx.x;
  long total = (long)Bsz*L*Hc; if(i >= total) return;
  int c = (int)(i % Hc); long t = i / Hc;
  int l = (int)(t % L);  int b = (int)(t / L);
  const float* xb = x + (long)b*(2*Hc)*L;
  o1[i] = xb[(long)(2*c)*L + l];
  o2[i] = xb[(long)(2*c+1)*L + l];
}

__global__ void k_ln_row(const float* __restrict__ x, const float* __restrict__ w,
                         const float* __restrict__ b, float* __restrict__ y,
                         int rows, int dim, float eps){
  long r = (long)blockIdx.x*blockDim.x + threadIdx.x;
  if(r >= rows) return;
  const float* xp = x + r*(long)dim;
  float mu = 0.0f;
  for(int i=0;i<dim;++i) mu += xp[i];
  mu /= dim;
  float var = 0.0f;
  for(int i=0;i<dim;++i){ float d = xp[i]-mu; var += d*d; }
  var /= dim;
  float inv = rsqrtf(var + eps);
  float* yp = y + r*(long)dim;
  for(int i=0;i<dim;++i) yp[i] = (xp[i]-mu)*inv*w[i] + b[i];
}

__global__ void k_ln_chan(const float* __restrict__ x, const float* __restrict__ w,
                          const float* __restrict__ b, float* __restrict__ y,
                          int Bsz, int C, int L){
  long i = (long)blockIdx.x*blockDim.x + threadIdx.x;
  if(i >= (long)Bsz*L) return;
  int l = (int)(i % L); int bb = (int)(i / L);
  const float* xp = x + (long)bb*C*L + l;
  float mu = 0.0f;
  for(int c=0;c<C;++c) mu += xp[(long)c*L];
  mu /= C;
  float var = 0.0f;
  for(int c=0;c<C;++c){ float d = xp[(long)c*L]-mu; var += d*d; }
  var /= C;
  float inv = rsqrtf(var + 1e-5f);
  float* yp = y + (long)bb*C*L + l;
  for(int c=0;c<C;++c) yp[(long)c*L] = (xp[(long)c*L]-mu)*inv*w[c] + b[c];
}

__global__ void k_swap(const float* __restrict__ x1, const float* __restrict__ x2,
                       float* __restrict__ s1, float* __restrict__ s2,
                       long rows, int dim){
  long i = (long)blockIdx.x*blockDim.x + threadIdx.x;
  if(i >= rows*(long)dim) return;
  int d = (int)(i % dim);
  int h = dim/2;
  if(d < h){ s1[i] = x2[i]; s2[i] = x1[i]; }
  else     { s1[i] = x1[i]; s2[i] = x2[i]; }
}

__global__ void k_negexp(const float* __restrict__ x, float* __restrict__ y, long n){
  long i = (long)blockIdx.x*blockDim.x + threadIdx.x;
  if(i < n) y[i] = -__expf(x[i]);
}

// y[b,d,l] = x[(b*L+l)*xstride + xoff + d]   (token-major -> channel-major)
__global__ void k_t_ld(const float* __restrict__ x, int xstride, int xoff,
                       float* __restrict__ y, int Bsz, int L, int Dd){
  long i = (long)blockIdx.x*blockDim.x + threadIdx.x;
  if(i >= (long)Bsz*Dd*L) return;
  int l = (int)(i % L); long t = i / L;
  int d = (int)(t % Dd); int b = (int)(t / Dd);
  y[i] = x[((long)b*L + l)*xstride + xoff + d];
}

__global__ void k_conv1d_silu(const float* __restrict__ x, const float* __restrict__ w,
                              const float* __restrict__ bias, float* __restrict__ y,
                              int Bsz, int Dd, int L){
  long i = (long)blockIdx.x*blockDim.x + threadIdx.x;
  if(i >= (long)Bsz*Dd*L) return;
  int l = (int)(i % L); long t = i / L;
  int d = (int)(t % Dd); int b = (int)(t / Dd);
  const float* xp = x + ((long)b*Dd + d)*L;
  float acc = bias[d];
#pragma unroll
  for(int j=0;j<4;++j){
    int idx = l - 3 + j;
    if(idx >= 0) acc += xp[idx]*w[d*4 + j];
  }
  y[i] = silu_(acc);
}

// out[b,l,d] = ys[b,d,l] * silu(z[(b*L+l)*zstride + zoff + d])
__global__ void k_mul_silu_t(const float* __restrict__ ys, const float* __restrict__ z,
                             int zstride, int zoff, float* __restrict__ out,
                             int Bsz, int L, int Dd){
  long i = (long)blockIdx.x*blockDim.x + threadIdx.x;
  if(i >= (long)Bsz*L*Dd) return;
  int d = (int)(i % Dd); long t = i / Dd;
  int l = (int)(t % L);  int b = (int)(t / L);
  float yv = ys[((long)b*Dd + d)*L + l];
  float zv = z[((long)b*L + l)*zstride + zoff + d];
  out[i] = yv * silu_(zv);
}

// y[r,d] *= silu(z[r*zstride + zoff + d])  (row-major, in place)
__global__ void k_mul_silu_rows(float* __restrict__ y, const float* __restrict__ z,
                                int zstride, int zoff, long rows, int dim){
  long i = (long)blockIdx.x*blockDim.x + threadIdx.x;
  if(i >= rows*(long)dim) return;
  int d = (int)(i % dim); long r = i / dim;
  y[i] *= silu_(z[r*zstride + zoff + d]);
}

__global__ void k_build_dirs(const float* __restrict__ xi, float* __restrict__ xs,
                             int Bsz, int Dd, int H, int W){
  long i = (long)blockIdx.x*blockDim.x + threadIdx.x;
  long L = (long)H*W;
  if(i >= (long)Bsz*Dd*L) return;
  int l = (int)(i % L); long t = i / L;
  int d = (int)(t % Dd); int b = (int)(t / Dd);
  int h = l / W, w = l % W;
  long lw = (long)w*H + h;
  float v = xi[i];
  float* p = xs + (long)b*4*Dd*L + (long)d*L;
  p[l] = v;
  p[(long)Dd*L + lw] = v;
  p[2L*(long)Dd*L + (L-1-l)] = v;
  p[3L*(long)Dd*L + (L-1-lw)] = v;
}

// out[b,l,d] (token-major) = sum of 4 direction scans
__global__ void k_combine(const float* __restrict__ y4, float* __restrict__ out,
                          int Bsz, int Dd, int H, int W){
  long i = (long)blockIdx.x*blockDim.x + threadIdx.x;
  long L = (long)H*W;
  if(i >= (long)Bsz*L*Dd) return;
  int d = (int)(i % Dd); long t = i / Dd;
  int l = (int)(t % L);  int b = (int)(t / L);
  int h = l / W, w = l % W;
  long lw = (long)w*H + h;
  const float* p = y4 + (long)b*4*Dd*L + (long)d*L;
  out[i] = p[l] + p[(long)Dd*L + lw] + p[2L*(long)Dd*L + (L-1-l)]
         + p[3L*(long)Dd*L + (L-1-lw)];
}

// x[r,d] = x[r,d]*s[d] + t[r,d]
__global__ void k_scale_add(float* __restrict__ x, const float* __restrict__ s,
                            const float* __restrict__ tt, long rows, int dim){
  long i = (long)blockIdx.x*blockDim.x + threadIdx.x;
  if(i >= rows*(long)dim) return;
  int d = (int)(i % dim);
  x[i] = x[i]*s[d] + tt[i];
}

// x_nhwc[b,l,c] = x*s[c] + t_nchw[b,c,l]
__global__ void k_scaleadd_nchw(float* __restrict__ x, const float* __restrict__ s,
                                const float* __restrict__ tt, int Bsz, int L, int C){
  long i = (long)blockIdx.x*blockDim.x + threadIdx.x;
  if(i >= (long)Bsz*L*C) return;
  int c = (int)(i % C); long t = i / C;
  int l = (int)(t % L); int b = (int)(t / L);
  x[i] = x[i]*s[c] + tt[((long)b*C + c)*L + l];
}

__global__ void k_mean_hw(const float* __restrict__ x, float* __restrict__ o,
                          int Bsz, int C, int L){
  long i = (long)blockIdx.x*blockDim.x + threadIdx.x;
  if(i >= (long)Bsz*C) return;
  const float* xp = x + i*(long)L;
  float s = 0.0f;
  for(int l=0;l<L;++l) s += xp[l];
  o[i] = s / L;
}

__global__ void k_fc_small(const float* __restrict__ x, const float* __restrict__ w,
                           const float* __restrict__ b, float* __restrict__ y,
                           int Bsz, int Cin, int Cout, int act){
  long i = (long)blockIdx.x*blockDim.x + threadIdx.x;
  if(i >= (long)Bsz*Cout) return;
  int o = (int)(i % Cout); int bb = (int)(i / Cout);
  float acc = b[o];
  for(int c=0;c<Cin;++c) acc += w[(long)o*Cin + c]*x[(long)bb*Cin + c];
  y[i] = actf(acc, act);
}

__global__ void k_mul_chan(float* __restrict__ y, const float* __restrict__ s,
                           int Bsz, int C, int L){
  long i = (long)blockIdx.x*blockDim.x + threadIdx.x;
  if(i >= (long)Bsz*C*L) return;
  long bc = i / L;
  y[i] *= s[bc];
}

__global__ void k_bn(const float* __restrict__ x, const float* __restrict__ w,
                     const float* __restrict__ b, const float* __restrict__ mean,
                     const float* __restrict__ var, float* __restrict__ y,
                     int Bsz, int C, int L){
  long i = (long)blockIdx.x*blockDim.x + threadIdx.x;
  if(i >= (long)Bsz*C*L) return;
  int c = (int)((i / L) % C);
  y[i] = (x[i]-mean[c])*rsqrtf(var[c]+1e-5f)*w[c] + b[c];
}

// Fmod[b,c,l] = (c<96 ? FaD[b,c,l] : F[b,c,l]) * gelu(F[b,192+c,l])
__global__ void k_gp_mod(const float* __restrict__ F, const float* __restrict__ FaD,
                         float* __restrict__ Fm, int Bsz, int L){
  long i = (long)blockIdx.x*blockDim.x + threadIdx.x;
  if(i >= (long)Bsz*192*L) return;
  int l = (int)(i % L); long t = i / L;
  int c = (int)(t % 192); int b = (int)(t / 192);
  const float* Fb = F + (long)b*384*L;
  float v = (c < 96) ? FaD[((long)b*96 + c)*L + l] : Fb[(long)c*L + l];
  Fm[i] = v * gelu_(Fb[(long)(192 + c)*L + l]);
}

__global__ void k_add(float* __restrict__ y, const float* __restrict__ x, long n){
  long i = (long)blockIdx.x*blockDim.x + threadIdx.x;
  if(i < n) y[i] += x[i];
}
__global__ void k_add_out(const float* __restrict__ a, const float* __restrict__ b,
                          float* __restrict__ o, long n){
  long i = (long)blockIdx.x*blockDim.x + threadIdx.x;
  if(i < n) o[i] = a[i] + b[i];
}

// out[b,c,l] = a[b*aB + c*L + l] * g[b*96L + c*L + l]
__global__ void k_mul_strided(const float* __restrict__ a, long aB,
                              const float* __restrict__ g, float* __restrict__ o,
                              int Bsz, int C, int L){
  long i = (long)blockIdx.x*blockDim.x + threadIdx.x;
  if(i >= (long)Bsz*C*L) return;
  int l = (int)(i % L); long t = i / L;
  int c = (int)(t % C); int b = (int)(t / C);
  o[i] = a[(long)b*aB + (long)c*L + l] * g[i];
}

// x1' = x1 + sigmoid(x2); x2' = x2 + sigmoid(x1'); xc = concat(x1',x2')
__global__ void k_cross(const float* __restrict__ x1, const float* __restrict__ x2,
                        float* __restrict__ xc, int Bsz, int L){
  long i = (long)blockIdx.x*blockDim.x + threadIdx.x;
  if(i >= (long)Bsz*96*L) return;
  int l = (int)(i % L); long t = i / L;
  int c = (int)(t % 96); int b = (int)(t / 96);
  float a  = x1[i];
  float b2 = x2[i];
  float a2 = a + sigmoid_(b2);
  float b3 = b2 + sigmoid_(a2);
  float* xb = xc + (long)b*192*L;
  xb[(long)c*L + l]        = a2;
  xb[(long)(96 + c)*L + l] = b3;
}

// g[b,c,l] = gelu(hd[b,c,l]) * hd[b,384+c,l]   (hd: 768 channels)
__global__ void k_gelu_mul(const float* __restrict__ hd, float* __restrict__ g,
                           int Bsz, int L){
  long i = (long)blockIdx.x*blockDim.x + threadIdx.x;
  if(i >= (long)Bsz*384*L) return;
  int l = (int)(i % L); long t = i / L;
  int c = (int)(t % 384); int b = (int)(t / 384);
  const float* hp = hd + (long)b*768*L;
  g[i] = gelu_(hp[(long)c*L + l]) * hp[(long)(384 + c)*L + l];
}

// ---------------------------------------------------------------------------
static inline dim3 g1d(long total){ return dim3((unsigned)((total + 255)/256)); }

extern "C" void kernel_launch(void* const* d_in, const int* in_sizes, int n_in,
                              void* d_out, int out_size, void* d_ws, size_t ws_size,
                              hipStream_t stream){
  (void)in_sizes; (void)out_size; (void)ws_size;

  const int  Bz = 2, Hh = 64, Wd = 64;
  const long L  = (long)Hh*Wd;           // 4096
  const int  DIMc = 96, DI = 192, NS = 16, RK = 6, CP = RK + 2*NS; // 38
  const long BL = (long)Bz*L;            // 8192

  const float* P[70];
  for(int i=0;i<n_in && i<70;++i) P[i] = (const float*)d_in[i];
  const float* d_x = P[0];

  // ---- workspace carve -----------------------------------------------------
  char* wsb = (char*)d_ws; size_t wsOff = 0;
  auto alloc = [&](long nElem)->float*{
    float* p = (float*)(wsb + wsOff);
    wsOff += (((size_t)nElem*sizeof(float) + 255) & ~(size_t)255);
    return p;
  };
  float* f_in1  = alloc(BL*DIMc);
  float* f_in2  = alloc(BL*DIMc);
  float* f_x1s  = alloc(BL*DIMc);
  float* f_x2s  = alloc(BL*DIMc);
  float* f_out1 = alloc(BL*DIMc);
  float* f_out2 = alloc(BL*DIMc);
  float* A1 = alloc((long)DI*NS);
  float* A2 = alloc((long)DI*NS);
  float* A4 = alloc(4L*DI*NS);
  float* m_xz   = alloc(BL*2*DI);
  float* m_xt   = alloc((long)Bz*DI*L);
  float* m_xc   = alloc((long)Bz*DI*L);
  float* m_xdbl = alloc((long)Bz*CP*L);
  float* m_delta= alloc((long)Bz*DI*L);
  float* m_ys   = alloc((long)Bz*DI*L);
  float* m_ym   = alloc(BL*DI);
  float* big1 = alloc((long)Bz*4*DI*L);      // xs4 / ffn hidden (768ch)
  float* big2 = alloc((long)Bz*4*DI*L);      // delta4 / ffn dw
  float* big3 = alloc((long)Bz*4*DI*L);      // y4 / ffn gated (384ch)
  float* xdbl4 = alloc((long)Bz*4*CP*L);
  float* x2c  = alloc((long)Bz*96*L);
  float* xnb  = alloc((long)Bz*96*L);
  float* Fbuf = alloc((long)Bz*384*L);
  float* t96  = alloc((long)Bz*96*L);
  float* xcbuf= alloc((long)Bz*192*L);
  float* cmean= alloc((long)Bz*96);
  float* s1b  = alloc((long)Bz*3);
  float* s2b  = alloc((long)Bz*96);
  float* tdmb = alloc(64);                   // TDM drain slot (unused downstream)

  auto gemm = [&](const float* A, const float* B, const float* bias, float* C,
                  int M, int N, int K, int transB, int biasMode, int act,
                  long sA, long sB, long sC, int zmodA, long biasStride, int Z){
    dim3 grid((N + 63)/64, (M + 63)/64, Z);
    k_gemm_wmma<<<grid, 256, 0, stream>>>(A, B, bias, C, M, N, K, transB,
                                          biasMode, act, sA, sB, sC, zmodA, biasStride);
  };

  // ---- TDM probe: async-tensor load of a weight tile (independent) ---------
  k_tdm_stage<<<dim3(1), 32, 0, stream>>>(P[63], tdmb);

  // ---- split even/odd channels, per-token LN, half-swap --------------------
  k_extract<<<g1d(BL*DIMc),256,0,stream>>>(d_x, f_in1, f_in2, Bz, (int)L, DIMc);
  k_ln_row<<<g1d(BL),256,0,stream>>>(f_in1, P[1], P[2], f_in1, (int)BL, DIMc, 1e-5f);
  k_ln_row<<<g1d(BL),256,0,stream>>>(f_in2, P[3], P[4], f_in2, (int)BL, DIMc, 1e-5f);
  k_swap<<<g1d(BL*DIMc),256,0,stream>>>(f_in1, f_in2, f_x1s, f_x2s, BL, DIMc);
  k_negexp<<<g1d((long)DI*NS),256,0,stream>>>(P[11], A1, (long)DI*NS);
  k_negexp<<<g1d((long)DI*NS),256,0,stream>>>(P[20], A2, (long)DI*NS);
  k_negexp<<<g1d(4L*DI*NS),256,0,stream>>>(P[35], A4, 4L*DI*NS);

  // ---- mamba1d (shared scratch) -------------------------------------------
  auto run_mamba = [&](const float* xin, int pi, const float* Aneg, float* outp){
    gemm(xin, P[pi], nullptr, m_xz, (int)BL, 2*DI, DIMc, 1, 0, 0, 0,0,0, 1, 0, 1);
    k_t_ld<<<g1d((long)Bz*DI*L),256,0,stream>>>(m_xz, 2*DI, 0, m_xt, Bz, (int)L, DI);
    k_conv1d_silu<<<g1d((long)Bz*DI*L),256,0,stream>>>(m_xt, P[pi+1], P[pi+2], m_xc, Bz, DI, (int)L);
    gemm(P[pi+3], m_xc, nullptr, m_xdbl, CP, (int)L, DI, 0, 0, 0,
         0, (long)DI*L, (long)CP*L, 1, 0, Bz);
    gemm(P[pi+4], m_xdbl, P[pi+5], m_delta, DI, (int)L, RK, 0, 2, 3,
         0, (long)CP*L, (long)DI*L, 1, 0, Bz);
    k_scan<<<g1d((long)Bz*DI),256,0,stream>>>(m_xc, m_delta, Aneg,
         m_xdbl + 6*L, m_xdbl + 22*L, P[pi+7], m_ys, Bz, 1, DI, (int)L, (long)CP*L);
    k_mul_silu_t<<<g1d((long)Bz*L*DI),256,0,stream>>>(m_ys, m_xz, 2*DI, DI, m_ym, Bz, (int)L, DI);
    gemm(m_ym, P[pi+8], nullptr, outp, (int)BL, DIMc, DI, 1, 0, 0, 0,0,0, 1, 0, 1);
  };
  run_mamba(f_x1s, 5,  A1, f_out1);   // enc1
  run_mamba(f_x2s, 14, A2, f_out2);   // enc2

  // ---- SS2D on x1 ----------------------------------------------------------
  float* f_ln = f_x1s;                // reuse
  k_ln_row<<<g1d(BL),256,0,stream>>>(f_out1, P[23], P[24], f_ln, (int)BL, DIMc, 1e-6f);
  gemm(f_ln, P[29], nullptr, m_xz, (int)BL, 2*DI, DIMc, 1, 0, 0, 0,0,0, 1, 0, 1);
  k_t_ld<<<g1d((long)Bz*DI*L),256,0,stream>>>(m_xz, 2*DI, 0, m_xt, Bz, (int)L, DI);
  k_conv3x3<<<g1d((long)Bz*DI*L),256,0,stream>>>(m_xt, P[30], P[31], m_xc,
      Bz, DI, DI, Hh, Wd, DI, 1, (long)DI*L, (long)DI*L);          // dw 3x3 + silu
  k_build_dirs<<<g1d((long)Bz*DI*L),256,0,stream>>>(m_xc, big1, Bz, DI, Hh, Wd);
  gemm(P[32], big1, nullptr, xdbl4, CP, (int)L, DI, 0, 0, 0,
       (long)CP*DI, (long)DI*L, (long)CP*L, 4, 0, Bz*4);
  gemm(P[33], xdbl4, P[34], big2, DI, (int)L, RK, 0, 2, 3,
       (long)DI*RK, (long)CP*L, (long)DI*L, 4, DI, Bz*4);
  k_scan<<<g1d((long)Bz*4*DI),256,0,stream>>>(big1, big2, A4,
       xdbl4 + 6*L, xdbl4 + 22*L, P[36], big3, Bz*4, 4, DI, (int)L, (long)CP*L);
  k_combine<<<g1d(BL*DI),256,0,stream>>>(big3, m_ym, Bz, DI, Hh, Wd);
  k_ln_row<<<g1d(BL),256,0,stream>>>(m_ym, P[37], P[38], m_ym, (int)BL, DI, 1e-5f);
  k_mul_silu_rows<<<g1d(BL*DI),256,0,stream>>>(m_ym, m_xz, 2*DI, DI, BL, DI);
  float* ssout = f_x2s;               // reuse
  gemm(m_ym, P[39], nullptr, ssout, (int)BL, DIMc, DI, 1, 0, 0, 0,0,0, 1, 0, 1);
  k_scale_add<<<g1d(BL*DIMc),256,0,stream>>>(f_out1, P[27], ssout, BL, DIMc);

  // ---- CAB on x1 -----------------------------------------------------------
  k_ln_row<<<g1d(BL),256,0,stream>>>(f_out1, P[25], P[26], f_ln, (int)BL, DIMc, 1e-6f);
  k_t_ld<<<g1d((long)Bz*96*L),256,0,stream>>>(f_ln, 96, 0, m_xt, Bz, (int)L, 96);
  k_conv3x3<<<g1d((long)Bz*32*L),256,0,stream>>>(m_xt, P[40], P[41], m_xc,
      Bz, 96, 32, Hh, Wd, 1, 2, (long)96*L, (long)32*L);           // c1 + gelu
  k_conv3x3<<<g1d((long)Bz*96*L),256,0,stream>>>(m_xc, P[42], P[43], m_delta,
      Bz, 32, 96, Hh, Wd, 1, 0, (long)32*L, (long)96*L);           // c2
  k_mean_hw<<<g1d((long)Bz*96),256,0,stream>>>(m_delta, cmean, Bz, 96, (int)L);
  k_fc_small<<<g1d((long)Bz*3),256,0,stream>>>(cmean, P[44], P[45], s1b, Bz, 96, 3, 4);
  k_fc_small<<<g1d((long)Bz*96),256,0,stream>>>(s1b, P[46], P[47], s2b, Bz, 3, 96, 5);
  k_mul_chan<<<g1d((long)Bz*96*L),256,0,stream>>>(m_delta, s2b, Bz, 96, (int)L);
  k_scaleadd_nchw<<<g1d(BL*DIMc),256,0,stream>>>(f_out1, P[28], m_delta, Bz, (int)L, DIMc);
  k_t_ld<<<g1d((long)Bz*96*L),256,0,stream>>>(f_out1, 96, 0, m_ys, Bz, (int)L, 96); // x1 NCHW

  // ---- gated_percept on x2 -------------------------------------------------
  k_t_ld<<<g1d((long)Bz*96*L),256,0,stream>>>(f_out2, 96, 0, x2c, Bz, (int)L, 96);
  k_bn<<<g1d((long)Bz*96*L),256,0,stream>>>(x2c, P[48], P[49], P[50], P[51], xnb, Bz, 96, (int)L);
  gemm(P[52], xnb, nullptr, Fbuf, 384, (int)L, 96, 0, 0, 0,
       0, (long)96*L, (long)384*L, 1, 0, Bz);
  k_conv3x3<<<g1d((long)Bz*96*L),256,0,stream>>>(Fbuf, P[53], nullptr, m_delta,
      Bz, 96, 96, Hh, Wd, 48, 0, (long)384*L, (long)96*L);         // grouped dconv on Fa
  k_gp_mod<<<g1d((long)Bz*192*L),256,0,stream>>>(Fbuf, m_delta, m_xc, Bz, (int)L);
  gemm(P[54], m_xc, nullptr, t96, 96, (int)L, 192, 0, 0, 0,
       0, (long)192*L, (long)96*L, 1, 0, Bz);
  k_add<<<g1d((long)Bz*96*L),256,0,stream>>>(t96, xnb, (long)Bz*96*L);
  gemm(P[55], t96, P[56], m_delta, 96, (int)L, 96, 0, 2, 0,
       0, (long)96*L, (long)96*L, 1, 0, Bz);                       // gp out (x2g)

  // ---- gated_ffn on x2 -----------------------------------------------------
  gemm(P[57], m_delta, P[58], Fbuf, 192, (int)L, 96, 0, 2, 2,
       0, (long)96*L, (long)192*L, 1, 0, Bz);                      // fn1 + gelu
  k_conv3x3<<<g1d((long)Bz*96*L),256,0,stream>>>(Fbuf + 96*L, P[59], P[60], t96,
      Bz, 96, 96, Hh, Wd, 96, 0, (long)192*L, (long)96*L);         // gate dw
  k_mul_strided<<<g1d((long)Bz*96*L),256,0,stream>>>(Fbuf, (long)192*L, t96, x2c, Bz, 96, (int)L);
  gemm(P[61], x2c, P[62], m_xt, 96, (int)L, 96, 0, 2, 2,
       0, (long)96*L, (long)96*L, 1, 0, Bz);                       // fn2 + gelu
  k_add<<<g1d((long)Bz*96*L),256,0,stream>>>(m_delta, m_xt, (long)Bz*96*L); // x2f

  // ---- cross-gating + neck + channel-LN + FFN ------------------------------
  k_cross<<<g1d((long)Bz*96*L),256,0,stream>>>(m_ys, m_delta, xcbuf, Bz, (int)L);
  gemm(P[63], xcbuf, P[64], m_xz, 192, (int)L, 192, 0, 2, 0,
       0, (long)192*L, (long)192*L, 1, 0, Bz);                     // neck
  k_add<<<g1d((long)Bz*192*L),256,0,stream>>>(m_xz, d_x, (long)Bz*192*L);
  k_ln_chan<<<g1d(BL),256,0,stream>>>(m_xz, P[65], P[66], xcbuf, Bz, 192, (int)L);
  gemm(P[67], xcbuf, nullptr, big1, 768, (int)L, 192, 0, 0, 0,
       0, (long)192*L, (long)768*L, 1, 0, Bz);                     // pin
  k_conv3x3<<<g1d((long)Bz*768*L),256,0,stream>>>(big1, P[68], nullptr, big2,
      Bz, 768, 768, Hh, Wd, 768, 0, (long)768*L, (long)768*L);     // dw
  k_gelu_mul<<<g1d((long)Bz*384*L),256,0,stream>>>(big2, big3, Bz, (int)L);
  gemm(P[69], big3, nullptr, m_xc, 192, (int)L, 384, 0, 0, 0,
       0, (long)384*L, (long)192*L, 1, 0, Bz);                     // pout
  k_add_out<<<g1d((long)Bz*192*L),256,0,stream>>>(m_xc, m_xz, (float*)d_out, (long)Bz*192*L);
}
